// MyModel_69612829933654
// MI455X (gfx1250) — matrix operand order repaired and checked by reading, not compiled, not run on hardware
//
#include <hip/hip_runtime.h>

// ---------------- problem constants ----------------
#define B_    2
#define S_    2048
#define H_    3072
#define NH_   32
#define NKV_  8
#define HD_   96
#define DQKV  ((NH_ + 2 * NKV_) * HD_)   // 4608
#define NEGF  (-1.0e9f)

typedef __attribute__((ext_vector_type(16))) __bf16 v16bf;
typedef __attribute__((ext_vector_type(8)))  float  v8f;
typedef __attribute__((ext_vector_type(4)))  int    v4i_t;

union FragBF { v16bf v; unsigned short h[16]; uint4 q[2]; };
union FragF  { v8f v; float f[8]; };

__device__ __forceinline__ unsigned short f2bf(float f) {
  union { float f; unsigned u; } x; x.f = f;
  unsigned u = x.u;
  unsigned r = (u + 0x7FFFu + ((u >> 16) & 1u)) >> 16;  // RNE
  return (unsigned short)r;
}
__device__ __forceinline__ float bf2f(unsigned short h) {
  union { unsigned u; float f; } x; x.u = ((unsigned)h) << 16;
  return x.f;
}

// ---------------- CDNA5 async global->LDS (guarded) ----------------
#if defined(__HIP_DEVICE_COMPILE__) && \
    __has_builtin(__builtin_amdgcn_global_load_async_to_lds_b128)
#define USE_ASYNC_LDS 1
#else
#define USE_ASYNC_LDS 0
#endif

#if USE_ASYNC_LDS
typedef __attribute__((address_space(1))) v4i_t* gv4i_p;  // global int4*
typedef __attribute__((address_space(3))) v4i_t* lv4i_p;  // LDS int4*
__device__ __forceinline__ void async_copy16(const void* g, void* l) {
  __builtin_amdgcn_global_load_async_to_lds_b128((gv4i_p)g, (lv4i_p)l, 0, 0);
}
#define ASYNC_WAIT() asm volatile("s_wait_asynccnt 0x0" ::: "memory")
#else
#define ASYNC_WAIT()
#endif

// =====================================================================
// WMMA GEMM: Out[M,N] = A[M,K] * B[N,K]^T
//   A: f32 or bf16 (template), B: f32 (converted to bf16 while staging)
//   Out: bf16 or f32 (template)
// Block: 256 threads (8 waves), tile 256x128, BK=32, double-buffered LDS.
// Wave grid 4x2: wave covers 64 rows x 64 cols (4x4 WMMA frags).
// =====================================================================
template <bool A_IS_BF16, bool OUT_BF16>
__global__ __launch_bounds__(256) void gemm_wmma(
    const void* __restrict__ Ap, const float* __restrict__ Bp,
    void* __restrict__ Outp, int M, int N, int K) {
  __shared__ __align__(16) unsigned short lA[2][256 * 40];  // 40-half row stride
  __shared__ __align__(16) unsigned short lB[2][128 * 40];

  const int tid  = threadIdx.x;
  const int wave = tid >> 5;
  const int lane = tid & 31;
  const int l16  = lane & 15;
  const int hi   = lane >> 4;
  const int m0 = blockIdx.y * 256;
  const int n0 = blockIdx.x * 128;
  const int wr = wave >> 1;  // 0..3
  const int wc = wave & 1;   // 0..1

  FragF acc[4][4];
  for (int mt = 0; mt < 4; mt++)
    for (int nt = 0; nt < 4; nt++)
      for (int r = 0; r < 8; r++) acc[mt][nt].f[r] = 0.0f;

  const int srA = tid >> 2;        // 0..63  (bf16 A path: 16B chunks)
  const int scA = (tid & 3) << 3;  // 0,8,16,24 halves
  const int sr  = tid >> 3;        // 0..31  (f32 convert path)
  const int sc  = (tid & 7) << 2;  // 0..28

  auto stageA = [&](int buf, int k0) {
    if constexpr (A_IS_BF16) {
      for (int i = 0; i < 4; i++) {
        int row = srA + (i << 6);
        const unsigned short* src =
            (const unsigned short*)Ap + (size_t)(m0 + row) * K + k0 + scA;
#if USE_ASYNC_LDS
        async_copy16(src, &lA[buf][row * 40 + scA]);
#else
        *(uint4*)(&lA[buf][row * 40 + scA]) = *(const uint4*)src;
#endif
      }
    } else {
      for (int i = 0; i < 8; i++) {
        int row = sr + (i << 5);
        float4 v =
            *(const float4*)((const float*)Ap + (size_t)(m0 + row) * K + k0 + sc);
        union { unsigned short h[4]; uint2 u; } t;
        t.h[0] = f2bf(v.x); t.h[1] = f2bf(v.y); t.h[2] = f2bf(v.z); t.h[3] = f2bf(v.w);
        *(uint2*)(&lA[buf][row * 40 + sc]) = t.u;
      }
    }
  };
  auto stageB = [&](int buf, int k0) {
    for (int i = 0; i < 4; i++) {
      int row = sr + (i << 5);
      float4 w = *(const float4*)(Bp + (size_t)(n0 + row) * K + k0 + sc);
      union { unsigned short h[4]; uint2 u; } t;
      t.h[0] = f2bf(w.x); t.h[1] = f2bf(w.y); t.h[2] = f2bf(w.z); t.h[3] = f2bf(w.w);
      *(uint2*)(&lB[buf][row * 40 + sc]) = t.u;
    }
  };

  stageA(0, 0);
  stageB(0, 0);
  if (A_IS_BF16) { ASYNC_WAIT(); }
  __syncthreads();

  int p = 0;
  for (int k0 = 0; k0 < K; k0 += 32) {
    const int kn = k0 + 32;
    // stage next tile into the other buffer (overlaps with WMMA below)
    if (kn < K) { stageA(p ^ 1, kn); stageB(p ^ 1, kn); }
    // prefetch two tiles ahead into L2
    if (k0 + 64 < K) {
      if constexpr (A_IS_BF16)
        __builtin_prefetch((const unsigned short*)Ap + (size_t)(m0 + srA) * K + k0 + 64 + scA, 0, 0);
      else
        __builtin_prefetch((const float*)Ap + (size_t)(m0 + sr) * K + k0 + 64 + sc, 0, 0);
      __builtin_prefetch(Bp + (size_t)(n0 + sr) * K + k0 + 64 + sc, 0, 0);
    }

    // fragments from buffer p (ISA 16-bit A 16x32 / B 32x16 lane layouts)
    FragBF af[4], bfr[4];
    const int aoff = hi * 8;
    for (int mt = 0; mt < 4; mt++) {
      int row = wr * 64 + mt * 16 + l16;
      af[mt].q[0] = *(const uint4*)(&lA[p][row * 40 + aoff]);
      af[mt].q[1] = *(const uint4*)(&lA[p][row * 40 + 16 + aoff]);
    }
    const int boff = hi * 8;
    for (int nt = 0; nt < 4; nt++) {
      int col = wc * 64 + nt * 16 + l16;
      bfr[nt].q[0] = *(const uint4*)(&lB[p][col * 40 + boff]);
      bfr[nt].q[1] = *(const uint4*)(&lB[p][col * 40 + boff + 16]);
    }
    for (int mt = 0; mt < 4; mt++)
      for (int nt = 0; nt < 4; nt++)
        acc[mt][nt].v = __builtin_amdgcn_wmma_f32_16x16x32_bf16(
            false, af[mt].v, false, bfr[nt].v, (short)0, acc[mt][nt].v, false, false);

    if (A_IS_BF16 && kn < K) { ASYNC_WAIT(); }
    __syncthreads();
    p ^= 1;
  }

  // writeback: C frag element r -> (row r + 8*hi, col l16)
  for (int mt = 0; mt < 4; mt++)
    for (int nt = 0; nt < 4; nt++)
      for (int r = 0; r < 8; r++) {
        int row = m0 + wr * 64 + mt * 16 + r + 8 * hi;
        int col = n0 + wc * 64 + nt * 16 + l16;
        if constexpr (OUT_BF16)
          ((unsigned short*)Outp)[(size_t)row * N + col] = f2bf(acc[mt][nt].f[r]);
        else
          ((float*)Outp)[(size_t)row * N + col] = acc[mt][nt].f[r];
      }
}

// =====================================================================
// RoPE + head split: qkv[B*S, 4608] bf16 -> qb/kb/vb [B, heads, S, HD] bf16
// =====================================================================
__global__ __launch_bounds__(256) void rope_split(
    const unsigned short* __restrict__ qkv, const float* __restrict__ cosb,
    const float* __restrict__ sinb, unsigned short* __restrict__ qb,
    unsigned short* __restrict__ kb, unsigned short* __restrict__ vb) {
  const int token = blockIdx.x;  // b*S + s
  const int b = token / S_, s = token % S_;
  const unsigned short* row = qkv + (size_t)token * DQKV;
  const float* cp = cosb + (size_t)token * HD_;
  const float* sp = sinb + (size_t)token * HD_;
  for (int e = threadIdx.x; e < DQKV; e += 256) {
    int head = e / HD_, d = e % HD_;
    float x = bf2f(row[e]);
    float out;
    if (head < NH_ + NKV_) {  // rope on q and k
      float c = cp[d], sn = sp[d];
      if (d < HD_ / 2) {
        float x2 = bf2f(row[head * HD_ + d + HD_ / 2]);
        out = x * c - x2 * sn;
      } else {
        float x1 = bf2f(row[head * HD_ + d - HD_ / 2]);
        out = x * c + x1 * sn;
      }
    } else {
      out = x;
    }
    unsigned short o = f2bf(out);
    if (head < NH_)
      qb[(((size_t)b * NH_ + head) * S_ + s) * HD_ + d] = o;
    else if (head < NH_ + NKV_)
      kb[(((size_t)b * NKV_ + (head - NH_)) * S_ + s) * HD_ + d] = o;
    else
      vb[(((size_t)b * NKV_ + (head - NH_ - NKV_)) * S_ + s) * HD_ + d] = o;
  }
}

// =====================================================================
// Flash attention (causal, GQA rep=4): block = (head, 128-row q tile),
// 8 waves x 16 q rows each; kv tiles of 64; online softmax in registers.
// =====================================================================
__global__ __launch_bounds__(256) void flash_attn(
    const unsigned short* __restrict__ qb, const unsigned short* __restrict__ kb,
    const unsigned short* __restrict__ vb, unsigned short* __restrict__ ctx) {
  __shared__ __align__(16) unsigned short Kl[64 * 96];      // [kv][d]
  __shared__ __align__(16) unsigned short Vt[96 * 72];      // [d][kv], stride 72
  __shared__ __align__(16) unsigned short Pl[8][16 * 72];   // per-wave P, stride 72

  const int tid = threadIdx.x, wave = tid >> 5, lane = tid & 31;
  const int l16 = lane & 15, hi = lane >> 4;
  const int bh = blockIdx.x;  // b*NH + h
  const int qi = blockIdx.y;  // q tile (128 rows)
  const int b = bh / NH_, h = bh % NH_, kh = h / (NH_ / NKV_);

  const unsigned short* qptr =
      qb + (((size_t)b * NH_ + h) * S_ + qi * 128 + wave * 16) * HD_;
  const unsigned short* kptr = kb + (((size_t)b * NKV_ + kh) * S_) * HD_;
  const unsigned short* vptr = vb + (((size_t)b * NKV_ + kh) * S_) * HD_;

  // Q fragments straight from global (A 16x32 layout: aligned 16B chunks)
  FragBF aq[3];
  {
    const unsigned short* qr = qptr + (size_t)l16 * HD_;
    const int off8 = hi * 8;
    for (int kc = 0; kc < 3; kc++) {
      aq[kc].q[0] = *(const uint4*)(qr + kc * 32 + off8);
      aq[kc].q[1] = *(const uint4*)(qr + kc * 32 + 16 + off8);
    }
  }

  float mstate[8], lstate[8];
  FragF o6[6];
  for (int r = 0; r < 8; r++) { mstate[r] = -1.0e30f; lstate[r] = 0.0f; }
  for (int nt = 0; nt < 6; nt++)
    for (int r = 0; r < 8; r++) o6[nt].f[r] = 0.0f;

  const float scale = 0.10206207261596575f;  // 1/sqrt(96)
  const float LOG2E = 1.4426950408889634f;
  const int rowg0 = qi * 128 + wave * 16 + 8 * hi;
  const int jmax = 2 * qi + 1;

  for (int j = 0; j <= jmax; j++) {
    // prefetch next kv tile into L2
    if (j < jmax) {
      __builtin_prefetch(kptr + (size_t)(j + 1) * 64 * HD_ + tid * 24, 0, 0);
      __builtin_prefetch(vptr + (size_t)(j + 1) * 64 * HD_ + tid * 24, 0, 0);
    }
    // ---- stage K tile (async copy) and V tile (transposed, VGPR path) ----
    {
      const unsigned short* ks = kptr + (size_t)j * 64 * HD_;
#if USE_ASYNC_LDS
      for (int i = 0; i < 3; i++) {
        int idx = (tid + i * 256) * 8;  // halves, 16B chunks
        async_copy16(ks + idx, &Kl[idx]);
      }
#else
      const uint4* ks4 = (const uint4*)ks;
      uint4* kd = (uint4*)Kl;
      for (int i = 0; i < 3; i++) kd[tid + i * 256] = ks4[tid + i * 256];
#endif
      const unsigned short* vs = vptr + (size_t)j * 64 * HD_;
      for (int i = 0; i < 3; i++) {
        int e0 = (tid + i * 256) * 8;
        int kv = e0 / 96, d0 = e0 % 96;
        uint4 vv = *(const uint4*)(vs + e0);
        const unsigned short* hh = (const unsigned short*)&vv;
        for (int q = 0; q < 8; q++) Vt[(d0 + q) * 72 + kv] = hh[q];
      }
    }
    ASYNC_WAIT();
    __syncthreads();

    // ---- scores: S = Q * K^T (4 col tiles x 3 K-chunks of HD) ----
    FragF sf[4];
    for (int nt = 0; nt < 4; nt++) {
      FragF s;
      for (int r = 0; r < 8; r++) s.f[r] = 0.0f;
      const int col = nt * 16 + l16;
      for (int kc = 0; kc < 3; kc++) {
        FragBF bk;
        int koff = kc * 32 + hi * 8;
        bk.q[0] = *(const uint4*)(&Kl[col * 96 + koff]);
        bk.q[1] = *(const uint4*)(&Kl[col * 96 + koff + 16]);
        s.v = __builtin_amdgcn_wmma_f32_16x16x32_bf16(
            false, aq[kc].v, false, bk.v, (short)0, s.v, false, false);
      }
      sf[nt] = s;
    }

    // ---- scale + causal mask ----
    for (int nt = 0; nt < 4; nt++) {
      int colg = j * 64 + nt * 16 + l16;
      for (int r = 0; r < 8; r++) {
        float v = sf[nt].f[r] * scale;
        sf[nt].f[r] = (colg > rowg0 + r) ? NEGF : v;
      }
    }

    // ---- online softmax: row max / rescale / exp / row sum ----
    float mnew[8], cf[8];
    for (int r = 0; r < 8; r++) {
      float pm = sf[0].f[r];
      for (int nt = 1; nt < 4; nt++) pm = fmaxf(pm, sf[nt].f[r]);
      pm = fmaxf(pm, __shfl_xor(pm, 1, 32));
      pm = fmaxf(pm, __shfl_xor(pm, 2, 32));
      pm = fmaxf(pm, __shfl_xor(pm, 4, 32));
      pm = fmaxf(pm, __shfl_xor(pm, 8, 32));
      mnew[r] = fmaxf(mstate[r], pm);
      cf[r] = exp2f((mstate[r] - mnew[r]) * LOG2E);
      mstate[r] = mnew[r];
    }
    for (int r = 0; r < 8; r++) {
      float rs = 0.0f;
      int rl = r + 8 * hi;
      for (int nt = 0; nt < 4; nt++) {
        float p = exp2f((sf[nt].f[r] - mnew[r]) * LOG2E);
        rs += p;
        Pl[wave][rl * 72 + nt * 16 + l16] = f2bf(p);
      }
      rs += __shfl_xor(rs, 1, 32);
      rs += __shfl_xor(rs, 2, 32);
      rs += __shfl_xor(rs, 4, 32);
      rs += __shfl_xor(rs, 8, 32);
      lstate[r] = lstate[r] * cf[r] + rs;
    }
    for (int nt = 0; nt < 6; nt++)
      for (int r = 0; r < 8; r++) o6[nt].f[r] *= cf[r];
    __syncthreads();

    // ---- O += P * V (2 K-chunks of 32 keys x 6 d tiles) ----
    for (int kc = 0; kc < 2; kc++) {
      FragBF ap;
      const unsigned short* pr = &Pl[wave][l16 * 72 + kc * 32];
      const int off8 = hi * 8;
      ap.q[0] = *(const uint4*)(pr + off8);
      ap.q[1] = *(const uint4*)(pr + 16 + off8);
      for (int nt = 0; nt < 6; nt++) {
        FragBF bv;
        int drow = nt * 16 + l16;
        int koff = kc * 32 + hi * 8;
        bv.q[0] = *(const uint4*)(&Vt[drow * 72 + koff]);
        bv.q[1] = *(const uint4*)(&Vt[drow * 72 + koff + 16]);
        o6[nt].v = __builtin_amdgcn_wmma_f32_16x16x32_bf16(
            false, ap.v, false, bv.v, (short)0, o6[nt].v, false, false);
      }
    }
    __syncthreads();
  }

  // ---- epilogue: normalize and store ctx bf16 [B*S, NH*HD] ----
  for (int nt = 0; nt < 6; nt++)
    for (int r = 0; r < 8; r++) {
      int sg = qi * 128 + wave * 16 + r + 8 * hi;
      int d = nt * 16 + l16;
      float v = o6[nt].f[r] / lstate[r];
      ctx[((size_t)b * S_ + sg) * (NH_ * HD_) + h * HD_ + d] = f2bf(v);
    }
}

// =====================================================================
extern "C" void kernel_launch(void* const* d_in, const int* in_sizes, int n_in,
                              void* d_out, int out_size, void* d_ws, size_t ws_size,
                              hipStream_t stream) {
  const float* hs   = (const float*)d_in[0];
  const float* cosb = (const float*)d_in[1];
  const float* sinb = (const float*)d_in[2];
  const float* wqkv = (const float*)d_in[4];
  const float* wo   = (const float*)d_in[5];
  float* out = (float*)d_out;

  // workspace layout (bf16 halves)
  unsigned short* qkvb = (unsigned short*)d_ws;                  // [4096, 4608]
  unsigned short* qb = qkvb + (size_t)(B_ * S_) * DQKV;          // [B,NH,S,HD]
  unsigned short* kb = qb + (size_t)B_ * NH_ * S_ * HD_;         // [B,NKV,S,HD]
  unsigned short* vb = kb + (size_t)B_ * NKV_ * S_ * HD_;        // [B,NKV,S,HD]
  unsigned short* ctx = qkvb;  // reuse: qkv dead after rope_split

  // 1) qkv = hs * Wqkv^T  (M=4096, N=4608, K=3072), bf16 out
  dim3 g1(DQKV / 128, (B_ * S_) / 256);
  gemm_wmma<false, true><<<g1, 256, 0, stream>>>(hs, wqkv, qkvb, B_ * S_, DQKV, H_);

  // 2) RoPE + split into per-head layouts
  rope_split<<<B_ * S_, 256, 0, stream>>>(qkvb, cosb, sinb, qb, kb, vb);

  // 3) causal flash attention -> ctx bf16
  dim3 g2(B_ * NH_, S_ / 128);
  flash_attn<<<g2, 256, 0, stream>>>(qb, kb, vb, ctx);

  // 4) out = ctx * Wo^T  (M=4096, N=3072, K=3072), f32 out
  dim3 g3(H_ / 128, (B_ * S_) / 256);
  gemm_wmma<true, false><<<g3, 256, 0, stream>>>(ctx, wo, out, B_ * S_, H_, H_);
}
